// DecoderRNN_30150670417947
// MI455X (gfx1250) — matrix operand order repaired and checked by reading, not compiled, not run on hardware
//
#include <hip/hip_runtime.h>
#include <math.h>

// ---------------- problem constants ----------------
#define Bsz   64
#define Pn    196
#define ENCn  2048
#define En    512
#define Hn    512
#define Vn    10000
#define An    512
#define Tn    20
#define TCAPn 21
#define XD    (En + ENCn)   // 2560

// ---------------- WMMA types ----------------
typedef __attribute__((ext_vector_type(16))) __bf16 v16bf;
typedef __attribute__((ext_vector_type(8)))  float  v8f;

union FragU { uint4 q[2]; v16bf v; };

__device__ __forceinline__ unsigned short f2bf(float f) {
  unsigned int u = __float_as_uint(f);
  u += 0x7FFFu + ((u >> 16) & 1u);   // round-to-nearest-even
  return (unsigned short)(u >> 16);
}
__device__ __forceinline__ unsigned int pack2(float a, float b) {
  return (unsigned int)f2bf(a) | ((unsigned int)f2bf(b) << 16);
}
__device__ __forceinline__ uint4 pack8(const float* p) {
  float4 x = *(const float4*)p;
  float4 y = *(const float4*)(p + 4);
  uint4 r;
  r.x = pack2(x.x, x.y); r.y = pack2(x.z, x.w);
  r.z = pack2(y.x, y.y); r.w = pack2(y.z, y.w);
  return r;
}

// ---------------- pre-conversion kernels (run once per launch) ----------------
__global__ void cvt_bf16_kernel(const float* __restrict__ in,
                                unsigned short* __restrict__ out, long n)
{
  long i = ((long)blockIdx.x * blockDim.x + threadIdx.x) * 8;
  if (i + 8 <= n) *(uint4*)(out + i) = pack8(in + i);
}

// transpose + convert: in f32 [K,N] (ldin) -> out bf16 [N,K]
#define TT 32
__global__ __launch_bounds__(256) void cvtT_bf16_kernel(
    const float* __restrict__ in, int ldin,
    unsigned short* __restrict__ out, int K, int N)
{
  __shared__ float tile[TT][TT + 1];
  const int kb = blockIdx.y * TT, nb = blockIdx.x * TT;
  const int tx = threadIdx.x & 31;
  const int ty = threadIdx.x >> 5;
  for (int kk = ty; kk < TT; kk += 8) {
    const int gk = kb + kk, gn = nb + tx;
    tile[kk][tx] = (gk < K && gn < N) ? in[(size_t)gk * ldin + gn] : 0.f;
  }
  __syncthreads();
  for (int nn = ty; nn < TT; nn += 8) {
    const int gn = nb + nn, gk = kb + tx;
    if (gn < N && gk < K) out[(size_t)gn * K + gk] = f2bf(tile[tx][nn]);
  }
}

// ---------------- bf16-WMMA GEMM, async-to-LDS double-buffered pipeline --------
// C[M,N](ldc, f32) = A(bf16 [M,K]) * B(bf16 [N,K])^T_n (+ bias[n]) (+ C)
// Assumptions: M % 64 == 0, K % 64 == 0. N may be ragged (row-granular zero-fill
// in staging + per-column guard in epilogue).
#define BM 64
#define BN 64
#define BK 64

template<bool ACC, bool BIAS>
__global__ __launch_bounds__(256) void gemm_wmma(
    const unsigned short* __restrict__ Ab,   // bf16 [M,K]
    const unsigned short* __restrict__ Bb,   // bf16 [N,K]
    const float* __restrict__ bias,
    float* __restrict__ C, long ldc,
    int M, int N, int K)
{
  __shared__ __align__(16) unsigned short lsA[2][BM * BK];
  __shared__ __align__(16) unsigned short lsB[2][BN * BK];

  const int tid  = threadIdx.x;
  const int lane = tid & 31;
  const int wave = tid >> 5;
  const int m0 = blockIdx.y * BM;
  const int n0 = blockIdx.x * BN;

  const int wm  = (wave & 3) * 16;
  const int wn  = (wave >> 2) * 32;
  const int r   = lane & 15;
  const int hlf = lane >> 4;

  v8f acc0 = {}; v8f acc1 = {};

  const int arow = tid >> 2;          // 0..63
  const int acol = (tid & 3) * 16;    // 0,16,32,48
  const int sidx = arow * BK + acol;

  const bool fullN = (n0 + BN <= N);
  const bool bok   = fullN || (n0 + arow) < N;

  const unsigned short* ap = Ab + (size_t)(m0 + arow) * K + acol;
  const unsigned short* bp = Bb + (size_t)(n0 + arow) * K + acol;

  // LDS byte addresses for async destinations (low 32 bits of generic pointer)
  unsigned la[2], lb[2];
  la[0] = (unsigned)(size_t)&lsA[0][sidx];
  la[1] = (unsigned)(size_t)&lsA[1][sidx];
  lb[0] = (unsigned)(size_t)&lsB[0][sidx];
  lb[1] = (unsigned)(size_t)&lsB[1][sidx];

  // ragged-N rows: zero-fill both buffers once; async never overwrites them
  if (!bok) {
    const uint4 z = make_uint4(0, 0, 0, 0);
    uint4* d0 = (uint4*)&lsB[0][sidx];
    uint4* d1 = (uint4*)&lsB[1][sidx];
    d0[0] = z; d0[1] = z; d1[0] = z; d1[1] = z;
  }

  // ---- prologue: async-stage tile 0 into buffer 0 ----
  asm volatile("global_load_async_to_lds_b128 %0, %1, off"
               :: "v"(la[0]), "v"(ap) : "memory");
  asm volatile("global_load_async_to_lds_b128 %0, %1, off offset:16"
               :: "v"(la[0]), "v"(ap) : "memory");
  if (bok) {
    asm volatile("global_load_async_to_lds_b128 %0, %1, off"
                 :: "v"(lb[0]), "v"(bp) : "memory");
    asm volatile("global_load_async_to_lds_b128 %0, %1, off offset:16"
                 :: "v"(lb[0]), "v"(bp) : "memory");
  }
  asm volatile("s_wait_asynccnt 0" ::: "memory");
  __syncthreads();

  int cur = 0;
  for (int k0 = 0; k0 < K; k0 += BK) {
    const bool more = (k0 + BK) < K;
    // ---- issue async copies for the NEXT tile while computing this one ----
    if (more) {
      const int nxt = cur ^ 1;
      const unsigned short* ga = ap + k0 + BK;
      asm volatile("global_load_async_to_lds_b128 %0, %1, off"
                   :: "v"(la[nxt]), "v"(ga) : "memory");
      asm volatile("global_load_async_to_lds_b128 %0, %1, off offset:16"
                   :: "v"(la[nxt]), "v"(ga) : "memory");
      if (bok) {
        const unsigned short* gb = bp + k0 + BK;
        asm volatile("global_load_async_to_lds_b128 %0, %1, off"
                     :: "v"(lb[nxt]), "v"(gb) : "memory");
        asm volatile("global_load_async_to_lds_b128 %0, %1, off offset:16"
                     :: "v"(lb[nxt]), "v"(gb) : "memory");
      }
    }
    // ---- compute current tile: 4 WMMAs ----
    const unsigned short* As = lsA[cur];
    const unsigned short* Bs = lsB[cur];
#pragma unroll
    for (int ks = 0; ks < BK; ks += 32) {
      FragU fa;
      {
        const unsigned short* p = As + (wm + r) * BK + ks + hlf * 16;
        fa.q[0] = *(const uint4*)(p);
        fa.q[1] = *(const uint4*)(p + 8);
      }
      FragU fb0, fb1;
      {
        const unsigned short* p0 = Bs + (wn + r) * BK + ks + hlf * 16;
        fb0.q[0] = *(const uint4*)(p0);
        fb0.q[1] = *(const uint4*)(p0 + 8);
        const unsigned short* p1 = Bs + (wn + 16 + r) * BK + ks + hlf * 16;
        fb1.q[0] = *(const uint4*)(p1);
        fb1.q[1] = *(const uint4*)(p1 + 8);
      }
      acc0 = __builtin_amdgcn_wmma_f32_16x16x32_bf16(false, fa.v, false, fb0.v,
                                                     (short)0, acc0, false, false);
      acc1 = __builtin_amdgcn_wmma_f32_16x16x32_bf16(false, fa.v, false, fb1.v,
                                                     (short)0, acc1, false, false);
    }
    if (more) {
      asm volatile("s_wait_asynccnt 0" ::: "memory");
      __syncthreads();
      cur ^= 1;
    }
  }

  // ---- epilogue ----
#pragma unroll
  for (int v = 0; v < 8; ++v) {
    const int row  = m0 + wm + hlf * 8 + v;   // always < M (M % 64 == 0)
    const int col0 = n0 + wn + r;
    const int col1 = col0 + 16;
    if (col0 < N) {
      float d = acc0[v];
      if (BIAS) d += bias[col0];
      if (ACC)  d += C[(size_t)row * ldc + col0];
      C[(size_t)row * ldc + col0] = d;
    }
    if (col1 < N) {
      float d = acc1[v];
      if (BIAS) d += bias[col1];
      if (ACC)  d += C[(size_t)row * ldc + col1];
      C[(size_t)row * ldc + col1] = d;
    }
  }
}

// ---------------- attention: scores + softmax + context(bf16 -> xcat) --------
__global__ __launch_bounds__(256) void attention_kernel(
    const float* __restrict__ att1,      // [B*P, A] f32
    const float* __restrict__ att2,      // [B, A]   f32
    const float* __restrict__ fullW,     // [A]
    const float* __restrict__ fullb,     // [1]
    const float* __restrict__ features,  // [B, P, ENC] f32
    unsigned short* __restrict__ xcat)   // bf16 [B, XD]; context at +En
{
  const int b = blockIdx.x;
  __shared__ float sE[Pn];

  const int tid = threadIdx.x, lane = tid & 31, wave = tid >> 5;

  const float* a2 = att2 + (size_t)b * An;
  for (int p = wave; p < Pn; p += 8) {
    const float* a1 = att1 + ((size_t)b * Pn + p) * An;
    float s = 0.f;
    for (int k = lane * 4; k < An; k += 128) {
      float4 x = *(const float4*)(a1 + k);
      float4 y = *(const float4*)(a2 + k);
      float4 w = *(const float4*)(fullW + k);
      float v;
      v = x.x + y.x; v = v > 0.f ? v : 0.f; s += v * w.x;
      v = x.y + y.y; v = v > 0.f ? v : 0.f; s += v * w.y;
      v = x.z + y.z; v = v > 0.f ? v : 0.f; s += v * w.z;
      v = x.w + y.w; v = v > 0.f ? v : 0.f; s += v * w.w;
    }
#pragma unroll
    for (int off = 16; off; off >>= 1) s += __shfl_down(s, off, 32);
    if (lane == 0) sE[p] = s + fullb[0];
  }
  __syncthreads();

  if (wave == 0) {
    float m = -1e30f;
    for (int p = lane; p < Pn; p += 32) m = fmaxf(m, sE[p]);
#pragma unroll
    for (int off = 16; off; off >>= 1) m = fmaxf(m, __shfl_down(m, off, 32));
    m = __shfl(m, 0, 32);
    float s = 0.f;
    for (int p = lane; p < Pn; p += 32) { float ex = __expf(sE[p] - m); sE[p] = ex; s += ex; }
#pragma unroll
    for (int off = 16; off; off >>= 1) s += __shfl_down(s, off, 32);
    s = __shfl(s, 0, 32);
    const float inv = 1.f / s;
    for (int p = lane; p < Pn; p += 32) sE[p] *= inv;
  }
  __syncthreads();

  const float* fbase = features + (size_t)b * Pn * ENCn;
  for (int e = tid * 4; e < ENCn; e += 1024) {
    float4 s = make_float4(0.f, 0.f, 0.f, 0.f);
    const float* fp = fbase + e;
    for (int p = 0; p < Pn; ++p) {
      float4 f = *(const float4*)(fp + (size_t)p * ENCn);
      const float a = sE[p];
      s.x += a * f.x; s.y += a * f.y; s.z += a * f.z; s.w += a * f.w;
    }
    uint2 pk; pk.x = pack2(s.x, s.y); pk.y = pack2(s.z, s.w);
    *(uint2*)&xcat[(size_t)b * XD + En + e] = pk;
  }
}

// ---------------- embedding gather (bf16) into xcat[:, 0:En] ----------------
__global__ void embed_kernel(const int* __restrict__ captions,
                             const float* __restrict__ embW,
                             unsigned short* __restrict__ xcat, int t)
{
  const int b = blockIdx.x;
  const int tok = captions[b * TCAPn + t];
  for (int e = threadIdx.x * 4; e < En; e += 1024) {
    float4 v = *(const float4*)&embW[(size_t)tok * En + e];
    uint2 pk; pk.x = pack2(v.x, v.y); pk.y = pack2(v.z, v.w);
    *(uint2*)&xcat[(size_t)b * XD + e] = pk;
  }
}

// ---------------- LSTM pointwise (h written as bf16) ----------------
__device__ __forceinline__ float sigmoidf(float x) { return 1.f / (1.f + __expf(-x)); }

__global__ void lstm_kernel(const float* __restrict__ gates,
                            unsigned short* __restrict__ h_bf,
                            float* __restrict__ c)
{
  const int idx = blockIdx.x * blockDim.x + threadIdx.x;
  if (idx >= Bsz * Hn) return;
  const int b = idx / Hn, j = idx % Hn;
  const float* g = gates + (size_t)b * 4 * Hn;
  const float ig = sigmoidf(g[j]);
  const float fg = sigmoidf(g[Hn + j]);
  const float gg = tanhf(g[2 * Hn + j]);
  const float og = sigmoidf(g[3 * Hn + j]);
  const float cn = fg * c[idx] + ig * gg;
  c[idx] = cn;
  h_bf[idx] = f2bf(og * tanhf(cn));
}

// ---------------- init: h_bf=0, c=0, bsum = bih + bhh ----------------
__global__ void init_state(const float* __restrict__ bih, const float* __restrict__ bhh,
                           float* __restrict__ bsum, unsigned short* __restrict__ h_bf,
                           float* __restrict__ c)
{
  const int idx = blockIdx.x * blockDim.x + threadIdx.x;
  if (idx < 4 * Hn) bsum[idx] = bih[idx] + bhh[idx];
  if (idx < Bsz * Hn) { h_bf[idx] = 0; c[idx] = 0.f; }
}

// ---------------- host ----------------
extern "C" void kernel_launch(void* const* d_in, const int* in_sizes, int n_in,
                              void* d_out, int out_size, void* d_ws, size_t ws_size,
                              hipStream_t stream)
{
  const float* features = (const float*)d_in[0];
  const int*   captions = (const int*)  d_in[1];
  const float* embW     = (const float*)d_in[2];
  const float* encW     = (const float*)d_in[3];
  const float* encb     = (const float*)d_in[4];
  const float* decW     = (const float*)d_in[5];
  const float* decb     = (const float*)d_in[6];
  const float* fullW    = (const float*)d_in[7];
  const float* fullb    = (const float*)d_in[8];
  const float* Wih      = (const float*)d_in[9];
  const float* Whh      = (const float*)d_in[10];
  const float* bih      = (const float*)d_in[11];
  const float* bhh      = (const float*)d_in[12];
  const float* fcW      = (const float*)d_in[13];
  const float* fcb      = (const float*)d_in[14];
  float* out = (float*)d_out;

  // -------- workspace layout: f32 region, then 16B-aligned bf16 region --------
  float* ws    = (float*)d_ws;
  float* att1  = ws;                                   // [B*P, A]
  float* att2  = att1 + (size_t)Bsz * Pn * An;         // [B, A]
  float* gates = att2 + (size_t)Bsz * An;              // [B, 4H]
  float* c     = gates + (size_t)Bsz * 4 * Hn;         // [B, H]
  float* bsum  = c + (size_t)Bsz * Hn;                 // [4H]

  unsigned short* h_bf   = (unsigned short*)(bsum + 4 * Hn);       // [B, H]
  unsigned short* xcat   = h_bf + (size_t)Bsz * Hn;                // [B, XD]
  unsigned short* Wih_bf = xcat + (size_t)Bsz * XD;                // [4H, XD]
  unsigned short* Whh_bf = Wih_bf + (size_t)4 * Hn * XD;           // [4H, H]
  unsigned short* encWT  = Whh_bf + (size_t)4 * Hn * Hn;           // [A, ENC]
  unsigned short* decWT  = encWT + (size_t)An * ENCn;              // [A, H]
  unsigned short* fcWT   = decWT + (size_t)An * Hn;                // [V, H]
  unsigned short* feat_bf= fcWT + (size_t)Vn * Hn;                 // [B*P, ENC]

  const dim3 blk(256);

  // -------- one-time conversions to bf16 --------
  cvt_bf16_kernel<<<dim3(((size_t)4 * Hn * XD / 8 + 255) / 256), blk, 0, stream>>>(
      Wih, Wih_bf, (long)4 * Hn * XD);
  cvt_bf16_kernel<<<dim3(((size_t)4 * Hn * Hn / 8 + 255) / 256), blk, 0, stream>>>(
      Whh, Whh_bf, (long)4 * Hn * Hn);
  cvt_bf16_kernel<<<dim3(((size_t)Bsz * Pn * ENCn / 8 + 255) / 256), blk, 0, stream>>>(
      features, feat_bf, (long)Bsz * Pn * ENCn);
  cvtT_bf16_kernel<<<dim3(An / TT, ENCn / TT), blk, 0, stream>>>(encW, An, encWT, ENCn, An);
  cvtT_bf16_kernel<<<dim3(An / TT, Hn / TT),   blk, 0, stream>>>(decW, An, decWT, Hn, An);
  cvtT_bf16_kernel<<<dim3((Vn + TT - 1) / TT, Hn / TT), blk, 0, stream>>>(fcW, Vn, fcWT, Hn, Vn);

  // att1 = features @ enc_att_W + enc_att_b   (M=12544, N=512, K=2048)
  gemm_wmma<false, true><<<dim3(An / BN, (Bsz * Pn) / BM), blk, 0, stream>>>(
      feat_bf, encWT, encb, att1, An, Bsz * Pn, An, ENCn);

  init_state<<<dim3((Bsz * Hn + 255) / 256), blk, 0, stream>>>(bih, bhh, bsum, h_bf, c);

  for (int t = 0; t < Tn; ++t) {
    // att2 = h @ dec_att_W + dec_att_b   (64 x 512, K=512)
    gemm_wmma<false, true><<<dim3(An / BN, 1), blk, 0, stream>>>(
        h_bf, decWT, decb, att2, An, Bsz, An, Hn);

    // scores + softmax + context -> xcat[:, En:] (bf16)
    attention_kernel<<<dim3(Bsz), blk, 0, stream>>>(att1, att2, fullW, fullb, features, xcat);

    // embedding -> xcat[:, :En] (bf16)
    embed_kernel<<<dim3(Bsz), blk, 0, stream>>>(captions, embW, xcat, t);

    // gates = xcat @ Wih^T + (bih + bhh)   (64 x 2048, K=2560)
    gemm_wmma<false, true><<<dim3((4 * Hn) / BN, 1), blk, 0, stream>>>(
        xcat, Wih_bf, bsum, gates, 4 * Hn, Bsz, 4 * Hn, XD);

    // gates += h @ Whh^T   (64 x 2048, K=512)
    gemm_wmma<true, false><<<dim3((4 * Hn) / BN, 1), blk, 0, stream>>>(
        h_bf, Whh_bf, nullptr, gates, 4 * Hn, Bsz, 4 * Hn, Hn);

    // pointwise LSTM: update h (bf16), c (f32)
    lstm_kernel<<<dim3((Bsz * Hn + 255) / 256), blk, 0, stream>>>(gates, h_bf, c);

    // logits = h @ fc_W + fc_b -> out[:, t, :]   (64 x 10000, K=512)
    gemm_wmma<false, true><<<dim3((Vn + BN - 1) / BN, 1), blk, 0, stream>>>(
        h_bf, fcWT, fcb, out + (size_t)t * Vn, (long)Tn * Vn, Bsz, Vn, Hn);
  }
}
